// SelfAttention_60790967108053
// MI455X (gfx1250) — matrix operand order repaired and checked by reading, not compiled, not run on hardware
//
#include <hip/hip_runtime.h>

typedef __attribute__((ext_vector_type(16))) _Float16 v16h;
typedef __attribute__((ext_vector_type(8)))  _Float16 v8h;
typedef __attribute__((ext_vector_type(8)))  float    v8f;

constexpr int kB = 4;
constexpr int kT = 2048;
constexpr int kC = 1024;
constexpr int kD = 128;
constexpr int kRows = kB * kT;   // 8192 total query rows

__device__ __forceinline__ v8f wmma16(v16h a, v16h b, v8f c) {
  // D = A(16x32 f16) * B(32x16 f16) + C(16x16 f32)
  return __builtin_amdgcn_wmma_f32_16x16x32_f16(
      /*neg_a=*/false, a, /*neg_b=*/false, b,
      /*c_mod=*/(short)0, c, /*reuse_a=*/false, /*reuse_b=*/false);
}

__device__ __forceinline__ v8f zero8() {
  v8f z;
  #pragma unroll
  for (int i = 0; i < 8; ++i) z[i] = 0.0f;
  return z;
}

// One wave-instruction async copy: each lane moves 16B global -> LDS.
__device__ __forceinline__ void async_copy16(unsigned lds_off, unsigned voff,
                                             const void* sbase) {
  asm volatile("global_load_async_to_lds_b128 %0, %1, %2"
               :: "v"(lds_off), "v"(voff), "s"(sbase) : "memory");
}

// ---------------------------------------------------------------------------
// Phase 0: f32 -> f16 conversion (scale folded in; Wq carries D^-0.5).
// ---------------------------------------------------------------------------
__global__ __launch_bounds__(256) void cvt_kernel(const float* __restrict__ src,
                                                  _Float16* __restrict__ dst,
                                                  int n, float scale) {
  const int i = (blockIdx.x * 256 + threadIdx.x) * 8;
  if (i >= n) return;
  const float4* s4 = (const float4*)(src + i);
  const float4 a = s4[0];
  const float4 b = s4[1];
  v8h o;
  o[0] = (_Float16)(a.x * scale); o[1] = (_Float16)(a.y * scale);
  o[2] = (_Float16)(a.z * scale); o[3] = (_Float16)(a.w * scale);
  o[4] = (_Float16)(b.x * scale); o[5] = (_Float16)(b.y * scale);
  o[6] = (_Float16)(b.z * scale); o[7] = (_Float16)(b.w * scale);
  *(v8h*)(dst + i) = o;
}

// ---------------------------------------------------------------------------
// Phase 1: QKV projection.  grid = (64, 3); block = 256 (8 waves).
// Double-buffered async-to-LDS staging of x and W K-chunks; 8 WMMA per chunk
// per wave.  Outputs q/v as f16 [B*T, D]; k transposed [B, D, T].
// ---------------------------------------------------------------------------
__global__ __launch_bounds__(256) void qkv_proj_kernel(
    const _Float16* __restrict__ xh,   // [kRows, kC] f16
    const _Float16* __restrict__ wh,   // [3, kC, kD] f16 (Wq pre-scaled)
    _Float16* __restrict__ qh,         // [kRows, kD]
    _Float16* __restrict__ kth,        // [kB, kD, kT]
    _Float16* __restrict__ vh)         // [kRows, kD]
{
  __shared__ _Float16 xs[2][128][32];   // ping-pong x tile, 8KB each
  __shared__ _Float16 ws[2][32][128];   // ping-pong W chunk, 8KB each

  const int tid  = threadIdx.x;
  const int wave = tid >> 5;
  const int lane = tid & 31;
  const int lrow = lane & 15;
  const int hf   = lane >> 4;
  const int row0 = blockIdx.x * 128;
  const int m    = blockIdx.y;          // 0=q, 1=k, 2=v
  const _Float16* W = wh + (size_t)m * (kC * kD);

  // Per-thread async segment descriptors (2 x-segments + 2 W-segments).
  unsigned xoff[2], xlds[2], woff[2], wlds[2];
  #pragma unroll
  for (int u = 0; u < 2; ++u) {
    const int s  = tid + u * 256;
    const int xr = s >> 2, xp = s & 3;          // x tile: 128 rows x 4 segs
    xoff[u] = (unsigned)(((row0 + xr) * kC + xp * 8) * 2);
    xlds[u] = (unsigned)(uintptr_t)&xs[0][xr][xp * 8];
    const int wr = s >> 4, wp = s & 15;         // W chunk: 32 rows x 16 segs
    woff[u] = (unsigned)((wr * kD + wp * 8) * 2);
    wlds[u] = (unsigned)(uintptr_t)&ws[0][wr][wp * 8];
  }

  v8f acc[8];
  #pragma unroll
  for (int n = 0; n < 8; ++n) acc[n] = zero8();

  // Issue chunk 0 into buffer 0 (4 async ops per wave per buffer).
  #pragma unroll
  for (int u = 0; u < 2; ++u) {
    async_copy16(xlds[u], xoff[u], xh);
    async_copy16(wlds[u], woff[u], W);
  }

  for (int kb = 0; kb < kC / 32; ++kb) {
    const int bb = kb & 1;
    if (kb + 1 < kC / 32) {
      // Issue next chunk into the other buffer, then wait for current 4 ops.
      #pragma unroll
      for (int u = 0; u < 2; ++u) {
        async_copy16(xlds[u] + (bb ^ 1) * 8192u, xoff[u] + (kb + 1) * 64u, xh);
        async_copy16(wlds[u] + (bb ^ 1) * 8192u, woff[u] + (kb + 1) * 8192u, W);
      }
      asm volatile("s_wait_asynccnt 0x4" ::: "memory");
    } else {
      asm volatile("s_wait_asynccnt 0x0" ::: "memory");
    }
    __syncthreads();

    // A operand (16x32 f16): lane holds row (lane&15), K chunks 8*hf, 16+8*hf.
    v16h a;
    {
      const v8h lo = *(const v8h*)&xs[bb][wave * 16 + lrow][8 * hf];
      const v8h hi = *(const v8h*)&xs[bb][wave * 16 + lrow][16 + 8 * hf];
      #pragma unroll
      for (int j = 0; j < 8; ++j) { a[j] = lo[j]; a[8 + j] = hi[j]; }
    }
    #pragma unroll
    for (int n = 0; n < 8; ++n) {
      v16h b;
      #pragma unroll
      for (int j = 0; j < 8; ++j) {
        b[j]     = ws[bb][8 * hf + j][16 * n + lrow];
        b[8 + j] = ws[bb][16 + 8 * hf + j][16 * n + lrow];
      }
      acc[n] = wmma16(a, b, acc[n]);
    }
    __syncthreads();   // all waves done reading bb before it is refilled
  }

  #pragma unroll
  for (int n = 0; n < 8; ++n) {
    #pragma unroll
    for (int v = 0; v < 8; ++v) {
      const int grow = row0 + wave * 16 + v + 8 * hf;  // C/D row = v + 8*hf
      const int col  = 16 * n + lrow;
      const _Float16 val = (_Float16)acc[n][v];
      if (m == 0) {
        qh[(size_t)grow * kD + col] = val;
      } else if (m == 2) {
        vh[(size_t)grow * kD + col] = val;
      } else {
        const int bidx = grow / kT;
        const int t    = grow % kT;
        kth[((size_t)bidx * kD + col) * kT + t] = val;  // transposed store
      }
    }
  }
}

// ---------------------------------------------------------------------------
// Phase 2: causal flash attention.  grid = 64; block = 256 (8 waves).
// One 16-query tile per wave; tiles strided across blocks for load balance.
// ---------------------------------------------------------------------------
__global__ __launch_bounds__(256) void flash_attn_kernel(
    const _Float16* __restrict__ qh,   // [kRows, kD] (pre-scaled)
    const _Float16* __restrict__ kth,  // [kB, kD, kT]
    const _Float16* __restrict__ vh,   // [kRows, kD]
    float* __restrict__ out)           // [kRows, kD] f32
{
  __shared__ _Float16 plds[8][16][32];  // per-wave P transpose buffer

  const int tid  = threadIdx.x;
  const int wave = tid >> 5;
  const int lane = tid & 31;
  const int lrow = lane & 15;
  const int hf   = lane >> 4;
  // Strided tile map: every block gets a uniform mix of causal-range lengths.
  const int g    = wave * 64 + blockIdx.x;   // query tile id (0..511)
  const int qg0  = g * 16;
  const int b    = qg0 / kT;
  const int qt   = qg0 % kT;

  // Load q tile (16 x 128) into 4 A-operand chunks.
  v16h qa[4];
  #pragma unroll
  for (int c = 0; c < 4; ++c) {
    const _Float16* p = qh + (size_t)(qg0 + lrow) * kD + 32 * c + 8 * hf;
    const v8h lo = *(const v8h*)p;
    const v8h hi = *(const v8h*)(p + 16);
    #pragma unroll
    for (int j = 0; j < 8; ++j) { qa[c][j] = lo[j]; qa[c][8 + j] = hi[j]; }
  }

  v8f acc[8];
  #pragma unroll
  for (int n = 0; n < 8; ++n) acc[n] = zero8();
  float mrun[8], lrun[8];
  #pragma unroll
  for (int v = 0; v < 8; ++v) { mrun[v] = -3.0e38f; lrun[v] = 0.0f; }

  const int nkb = (qt + 15) / 32 + 1;  // causal key-block count
  for (int kb = 0; kb < nkb; ++kb) {
    const int key0 = kb * 32;

    // S = q . k^T : two 16-key sub-tiles, 4 head-dim K-chunks each.
    v8f s0 = zero8(), s1 = zero8();
    #pragma unroll
    for (int c = 0; c < 4; ++c) {
      const _Float16* kp = kth + ((size_t)b * kD + 32 * c) * kT + key0 + lrow;
      if (kb + 1 < nkb) __builtin_prefetch(kp + 32, 0, 0);
      v16h b0, b1;
      #pragma unroll
      for (int j = 0; j < 8; ++j) {
        b0[j]     = kp[(size_t)(8 * hf + j) * kT];
        b0[8 + j] = kp[(size_t)(16 + 8 * hf + j) * kT];
        b1[j]     = kp[(size_t)(8 * hf + j) * kT + 16];
        b1[8 + j] = kp[(size_t)(16 + 8 * hf + j) * kT + 16];
      }
      s0 = wmma16(qa[c], b0, s0);
      s1 = wmma16(qa[c], b1, s1);
    }

    // Causal mask (row = qt + v + 8*hf, col = key0 [+16] + lrow).
    #pragma unroll
    for (int v = 0; v < 8; ++v) {
      const int qrow = qt + v + 8 * hf;
      if (key0 + lrow > qrow)      s0[v] = -INFINITY;
      if (key0 + 16 + lrow > qrow) s1[v] = -INFINITY;
    }

    // Row max across the 16 lanes of each half-wave.
    float mnew[8], alpha[8];
    #pragma unroll
    for (int v = 0; v < 8; ++v) {
      float r = fmaxf(s0[v], s1[v]);
      r = fmaxf(r, __shfl_xor(r, 1, 32));
      r = fmaxf(r, __shfl_xor(r, 2, 32));
      r = fmaxf(r, __shfl_xor(r, 4, 32));
      r = fmaxf(r, __shfl_xor(r, 8, 32));
      mnew[v]  = fmaxf(mrun[v], r);
      alpha[v] = __expf(mrun[v] - mnew[v]);
    }

    // p = exp(s - m); row sum; update running stats.
    v8f p0, p1;
    #pragma unroll
    for (int v = 0; v < 8; ++v) {
      p0[v] = __expf(s0[v] - mnew[v]);
      p1[v] = __expf(s1[v] - mnew[v]);
      float rs = p0[v] + p1[v];
      rs += __shfl_xor(rs, 1, 32);
      rs += __shfl_xor(rs, 2, 32);
      rs += __shfl_xor(rs, 4, 32);
      rs += __shfl_xor(rs, 8, 32);
      lrun[v] = lrun[v] * alpha[v] + rs;
      mrun[v] = mnew[v];
    }
    #pragma unroll
    for (int n = 0; n < 8; ++n)
      #pragma unroll
      for (int v = 0; v < 8; ++v) acc[n][v] *= alpha[v];

    // C-layout -> A-layout transpose of P through per-wave LDS.
    #pragma unroll
    for (int v = 0; v < 8; ++v) {
      plds[wave][v + 8 * hf][lrow]      = (_Float16)p0[v];
      plds[wave][v + 8 * hf][16 + lrow] = (_Float16)p1[v];
    }
    v16h pa;
    {
      const v8h lo = *(const v8h*)&plds[wave][lrow][8 * hf];
      const v8h hi = *(const v8h*)&plds[wave][lrow][16 + 8 * hf];
      #pragma unroll
      for (int j = 0; j < 8; ++j) { pa[j] = lo[j]; pa[8 + j] = hi[j]; }
    }

    // acc += P(16x32) . V(32x16) over 8 dim tiles.
    #pragma unroll
    for (int n = 0; n < 8; ++n) {
      const _Float16* vp = vh + (size_t)(b * kT + key0) * kD + 16 * n + lrow;
      v16h vb;
      #pragma unroll
      for (int j = 0; j < 8; ++j) {
        vb[j]     = vp[(size_t)(8 * hf + j) * kD];
        vb[8 + j] = vp[(size_t)(16 + 8 * hf + j) * kD];
      }
      acc[n] = wmma16(pa, vb, acc[n]);
    }
  }

  // Normalize and store f32 output.
  #pragma unroll
  for (int n = 0; n < 8; ++n) {
    #pragma unroll
    for (int v = 0; v < 8; ++v) {
      const int rg = qg0 + v + 8 * hf;
      out[(size_t)rg * kD + 16 * n + lrow] = acc[n][v] / lrun[v];
    }
  }
}

// ---------------------------------------------------------------------------
extern "C" void kernel_launch(void* const* d_in, const int* in_sizes, int n_in,
                              void* d_out, int out_size, void* d_ws, size_t ws_size,
                              hipStream_t stream) {
  const float* x  = (const float*)d_in[0];
  const float* Wq = (const float*)d_in[1];
  const float* Wk = (const float*)d_in[2];
  const float* Wv = (const float*)d_in[3];

  _Float16* xh  = (_Float16*)d_ws;                  // 16 MB
  _Float16* wh  = xh  + (size_t)kRows * kC;         // 768 KB, [3, C, D]
  _Float16* qh  = wh  + 3 * (size_t)kC * kD;        // 2 MB
  _Float16* kth = qh  + (size_t)kRows * kD;         // 2 MB, [B, D, T]
  _Float16* vh  = kth + (size_t)kB * kD * kT;       // 2 MB
  float* out = (float*)d_out;

  const float qscale = 0.08838834764831845f;        // D^-0.5 folded into Wq
  cvt_kernel<<<kRows * kC / 2048, 256, 0, stream>>>(x, xh, kRows * kC, 1.0f);
  cvt_kernel<<<kC * kD / 2048, 256, 0, stream>>>(Wq, wh, kC * kD, qscale);
  cvt_kernel<<<kC * kD / 2048, 256, 0, stream>>>(Wk, wh + (size_t)kC * kD, kC * kD, 1.0f);
  cvt_kernel<<<kC * kD / 2048, 256, 0, stream>>>(Wv, wh + 2 * (size_t)kC * kD, kC * kD, 1.0f);

  dim3 gproj(kRows / 128, 3);
  qkv_proj_kernel<<<gproj, 256, 0, stream>>>(xh, wh, qh, kth, vh);
  flash_attn_kernel<<<kRows / (16 * 8), 256, 0, stream>>>(qh, kth, vh, out);
}
